// LSHSoftmax_30588757082169
// MI455X (gfx1250) — compile-verified, matrix operand order checked by reading, not run on hardware
//
#include <hip/hip_runtime.h>
#include <hip/hip_bf16.h>
#include <stdint.h>

// ---------------------------------------------------------------------------
// LSH sampled-softmax logits: out[b,s] = dot(inputs[b,:], weight[sample_ids[s],:]) + bias[sample_ids[s]]
// B=1024, S=65536, D=128, fp32 in/out.
// Strategy: gather+split fp32 -> bf16(hi)+bf16(lo), then 3-term bf16 WMMA GEMM
// (hi*hi + hi*lo + lo*hi) for ~fp32 accuracy at bf16 matrix-core throughput.
// ---------------------------------------------------------------------------

#define D_FEAT    128
#define BATCH     1024
#define N_SAMPLES 65536

typedef __attribute__((ext_vector_type(16))) __bf16        v16bf;
typedef __attribute__((ext_vector_type(8)))  float         v8f;
typedef __attribute__((ext_vector_type(8)))  unsigned int  v8u;
typedef __attribute__((ext_vector_type(4)))  unsigned int  v4u;

__device__ __forceinline__ unsigned short f32_to_bf16_rne(float f) {
    unsigned int u = __builtin_bit_cast(unsigned int, f);
    unsigned int r = u + 0x7FFFu + ((u >> 16) & 1u);   // round-to-nearest-even
    return (unsigned short)(r >> 16);
}
__device__ __forceinline__ float bf16_bits_to_f32(unsigned short h) {
    unsigned int u = ((unsigned int)h) << 16;
    return __builtin_bit_cast(float, u);
}

// ---------------------------------------------------------------------------
// Prep 1: gather sampled weight rows + bias, split each fp32 into bf16 hi/lo.
// One thread per (s, d) element.
// ---------------------------------------------------------------------------
__global__ void __launch_bounds__(256)
lsh_prep_gather(const int* __restrict__ sample_ids,
                const float* __restrict__ weight,
                const float* __restrict__ bias,
                unsigned short* __restrict__ w_hi,
                unsigned short* __restrict__ w_lo,
                float* __restrict__ bias_g) {
    int t = blockIdx.x * blockDim.x + threadIdx.x;   // [0, S*D)
    int s = t >> 7;                                  // / 128
    int d = t & (D_FEAT - 1);
    int sid = sample_ids[s];
    float w = weight[(long long)sid * D_FEAT + d];
    unsigned short hb = f32_to_bf16_rne(w);
    float lo = w - bf16_bits_to_f32(hb);
    w_hi[t] = hb;
    w_lo[t] = f32_to_bf16_rne(lo);
    if (d == 0) bias_g[s] = bias[sid];
}

// ---------------------------------------------------------------------------
// Prep 2: split inputs into bf16 hi/lo. One thread per element.
// ---------------------------------------------------------------------------
__global__ void __launch_bounds__(256)
lsh_prep_inputs(const float* __restrict__ x,
                unsigned short* __restrict__ x_hi,
                unsigned short* __restrict__ x_lo) {
    int t = blockIdx.x * blockDim.x + threadIdx.x;   // [0, BATCH*D)
    float v = x[t];
    unsigned short hb = f32_to_bf16_rne(v);
    x_hi[t] = hb;
    x_lo[t] = f32_to_bf16_rne(v - bf16_bits_to_f32(hb));
}

// ---------------------------------------------------------------------------
// Fragment loader for V_WMMA_F32_16X16X32_BF16.
// 16-bit A (16x32) / B (32x16) layout (ISA 7.12.2): lane = half*16 + l.
//   half==0: VGPR0..3 hold K=0..7 (packed pairs), VGPR4..7 hold K=16..23
//   half==1: VGPR0..3 hold K=8..15,            VGPR4..7 hold K=24..31
// With K contiguous in memory this is two 16B loads at byte offsets
// half*16 and 32+half*16 from the row's K-window start.
// rowptr = &buf[row*128 + kstep*32]  (bf16 elements)
// ---------------------------------------------------------------------------
__device__ __forceinline__ v16bf load_frag(const unsigned short* rowptr, int half) {
    const char* p = (const char*)rowptr;
    v4u lo = *(const v4u*)(p + half * 16);
    v4u hi = *(const v4u*)(p + 32 + half * 16);
    v8u d;
    d[0] = lo[0]; d[1] = lo[1]; d[2] = lo[2]; d[3] = lo[3];
    d[4] = hi[0]; d[5] = hi[1]; d[6] = hi[2]; d[7] = hi[3];
    return __builtin_bit_cast(v16bf, d);
}

__device__ __forceinline__ v8f wmma_bf16(v16bf a, v16bf b, v8f c) {
    return __builtin_amdgcn_wmma_f32_16x16x32_bf16(
        /*neg_a=*/false, a, /*neg_b=*/false, b,
        /*c_mod=*/(short)0, c, /*reuse_a=*/false, /*reuse_b=*/false);
}

// ---------------------------------------------------------------------------
// Main GEMM: out[m, n] = sum_k x[m,k] * w[n,k] + bias_g[n]
// Wave tile: 32(M) x 64(N) = 2x4 fragments of 16x16.
// Block: 8 waves arranged 2(M) x 4(N) -> 64(M) x 256(N) per block.
// Grid: (S/256, BATCH/64) = (256, 16).
// 3-term bf16 emulation: acc += a_hi*b_hi + a_hi*b_lo + a_lo*b_hi.
// ---------------------------------------------------------------------------
__global__ void __launch_bounds__(256)
lsh_logits_wmma(const unsigned short* __restrict__ x_hi,
                const unsigned short* __restrict__ x_lo,
                const unsigned short* __restrict__ w_hi,
                const unsigned short* __restrict__ w_lo,
                const float* __restrict__ bias_g,
                float* __restrict__ out) {
    const int lane = threadIdx.x & 31;
    const int wid  = threadIdx.x >> 5;
    const int l    = lane & 15;
    const int half = lane >> 4;

    const int m_base = blockIdx.y * 64  + (wid >> 2) * 32;   // wave's first M row
    const int n_base = blockIdx.x * 256 + (wid & 3) * 64;    // wave's first N col

    v8f acc[2][4];
#pragma unroll
    for (int mt = 0; mt < 2; ++mt)
#pragma unroll
        for (int nt = 0; nt < 4; ++nt)
            acc[mt][nt] = v8f{0.f, 0.f, 0.f, 0.f, 0.f, 0.f, 0.f, 0.f};

#pragma unroll
    for (int ks = 0; ks < D_FEAT / 32; ++ks) {
        const int k0 = ks * 32;
        // A fragments for the wave's two 16-row M tiles (hi and lo parts)
        const long long arow0 = (long long)(m_base + l) * D_FEAT + k0;
        const long long arow1 = (long long)(m_base + 16 + l) * D_FEAT + k0;
        v16bf a_hi0 = load_frag(x_hi + arow0, half);
        v16bf a_lo0 = load_frag(x_lo + arow0, half);
        v16bf a_hi1 = load_frag(x_hi + arow1, half);
        v16bf a_lo1 = load_frag(x_lo + arow1, half);

#pragma unroll
        for (int nt = 0; nt < 4; ++nt) {
            const long long brow = (long long)(n_base + nt * 16 + l) * D_FEAT + k0;
            v16bf b_hi = load_frag(w_hi + brow, half);
            v16bf b_lo = load_frag(w_lo + brow, half);

            acc[0][nt] = wmma_bf16(a_hi0, b_hi, acc[0][nt]);
            acc[0][nt] = wmma_bf16(a_hi0, b_lo, acc[0][nt]);
            acc[0][nt] = wmma_bf16(a_lo0, b_hi, acc[0][nt]);

            acc[1][nt] = wmma_bf16(a_hi1, b_hi, acc[1][nt]);
            acc[1][nt] = wmma_bf16(a_hi1, b_lo, acc[1][nt]);
            acc[1][nt] = wmma_bf16(a_lo1, b_hi, acc[1][nt]);
        }
    }

    // Epilogue: C/D layout: lane half*16+l holds column n=l, rows v + half*8.
#pragma unroll
    for (int nt = 0; nt < 4; ++nt) {
        const int n = n_base + nt * 16 + l;
        const float bv = bias_g[n];
#pragma unroll
        for (int mt = 0; mt < 2; ++mt) {
#pragma unroll
            for (int v = 0; v < 8; ++v) {
                const int row = m_base + mt * 16 + v + half * 8;
                out[(long long)row * N_SAMPLES + n] = acc[mt][nt][v] + bv;
            }
        }
    }
}

// ---------------------------------------------------------------------------
// Launch
// ---------------------------------------------------------------------------
extern "C" void kernel_launch(void* const* d_in, const int* in_sizes, int n_in,
                              void* d_out, int out_size, void* d_ws, size_t ws_size,
                              hipStream_t stream) {
    const float* inputs     = (const float*)d_in[0];
    // d_in[1] = labels (unused by the forward logits computation)
    const int*   sample_ids = (const int*)d_in[2];
    const float* weight     = (const float*)d_in[3];
    const float* bias       = (const float*)d_in[4];
    float*       out        = (float*)d_out;

    // Workspace layout (all 256B-aligned):
    //   w_hi : S*D bf16  (16 MB)
    //   w_lo : S*D bf16  (16 MB)
    //   x_hi : B*D bf16  (256 KB)
    //   x_lo : B*D bf16  (256 KB)
    //   bias_g : S fp32  (256 KB)
    char* ws = (char*)d_ws;
    unsigned short* w_hi = (unsigned short*)(ws);
    unsigned short* w_lo = (unsigned short*)(ws + (size_t)N_SAMPLES * D_FEAT * 2);
    unsigned short* x_hi = (unsigned short*)(ws + (size_t)N_SAMPLES * D_FEAT * 4);
    unsigned short* x_lo = (unsigned short*)(ws + (size_t)N_SAMPLES * D_FEAT * 4
                                                + (size_t)BATCH * D_FEAT * 2);
    float* bias_g = (float*)(ws + (size_t)N_SAMPLES * D_FEAT * 4
                                + (size_t)BATCH * D_FEAT * 4);

    // Prep: gather + precision split
    lsh_prep_gather<<<(N_SAMPLES * D_FEAT) / 256, 256, 0, stream>>>(
        sample_ids, weight, bias, w_hi, w_lo, bias_g);
    lsh_prep_inputs<<<(BATCH * D_FEAT) / 256, 256, 0, stream>>>(
        inputs, x_hi, x_lo);

    // GEMM: 64M x 256N per block
    dim3 grid(N_SAMPLES / 256, BATCH / 64);
    lsh_logits_wmma<<<grid, 256, 0, stream>>>(x_hi, x_lo, w_hi, w_lo, bias_g, out);
}
